// DirectionalGraphAttention_56573309223524
// MI455X (gfx1250) — compile-verified, matrix operand order checked by reading, not run on hardware
//
#include <hip/hip_runtime.h>
#include <hip/hip_bf16.h>

typedef __attribute__((ext_vector_type(16))) _Float16 v16h;
typedef __attribute__((ext_vector_type(8)))  float    v8f;

#define CCH 128
#define HEADS 8
#define HD 16
#define OUT_SCALE 4096.0f
#define OUT_INV_SCALE (1.0f/4096.0f)

// ---------------- prep: fp32 x -> fp16 x16, and per-node channel mean ----------------
__global__ void prep_x_kernel(const float* __restrict__ x, _Float16* __restrict__ x16,
                              float* __restrict__ xm, int N) {
    __shared__ float sh[CCH];
    int n = blockIdx.x;
    int c = threadIdx.x;
    float v = x[(size_t)n * CCH + c];
    x16[(size_t)n * CCH + c] = (_Float16)v;
    sh[c] = v;
    __syncthreads();
    for (int s = 64; s > 0; s >>= 1) {
        if (c < s) sh[c] += sh[c + s];
        __syncthreads();
    }
    if (c == 0) xm[n] = sh[0] * (1.0f / (float)CCH);
}

// ---------------- convert weights to fp16 ----------------
__global__ void conv_w_kernel(const float* __restrict__ Wq, const float* __restrict__ Wk,
                              const float* __restrict__ Wv, const float* __restrict__ Wo,
                              _Float16* __restrict__ w16qkv, _Float16* __restrict__ wo16) {
    int t = blockIdx.x * blockDim.x + threadIdx.x;
    const int M = CCH * CCH;
    if (t < M)            w16qkv[t]           = (_Float16)Wq[t];
    else if (t < 2 * M)   w16qkv[t]           = (_Float16)Wk[t - M];
    else if (t < 3 * M)   w16qkv[t]           = (_Float16)Wv[t - 2 * M];
    else if (t < 4 * M)   wo16[t - 3 * M]     = (_Float16)Wo[t - 3 * M];
}

// ---------------- WMMA fragment loaders (16x32 A / 32x16 B, fp16) ----------------
// Lane<16: row=lane,   K in {kb+0..7, kb+16..23}
// Lane>=16: row=lane-16, K in {kb+8..15, kb+24..31}
__device__ __forceinline__ v16h load_frag_f16(const _Float16* __restrict__ base, int rowc,
                                              int kb, int khalf) {
    const _Float16* p = base + (size_t)rowc * CCH + kb + khalf * 8;
    v16h f;
#pragma unroll
    for (int j = 0; j < 8; ++j) { f[j] = p[j]; f[j + 8] = p[16 + j]; }
    return f;
}

__device__ __forceinline__ v16h load_frag_f32(const float* __restrict__ base, int rowc,
                                              int kb, int khalf, float scale) {
    const float* p = base + (size_t)rowc * CCH + kb + khalf * 8;
    v16h f;
#pragma unroll
    for (int j = 0; j < 8; ++j) {
        f[j]     = (_Float16)(p[j] * scale);
        f[j + 8] = (_Float16)(p[16 + j] * scale);
    }
    return f;
}

// ---------------- fused Q/K/V projection GEMM (x @ W.T + b) via WMMA ----------------
// block = 256 threads = 8 waves; wave w computes output tile (mBase, 16*w)
// blockIdx.z selects which of {Wq,Wk,Wv}
__global__ void gemm_qkv_kernel(const _Float16* __restrict__ x16,
                                const _Float16* __restrict__ w16qkv,
                                const float* __restrict__ bq, const float* __restrict__ bk,
                                const float* __restrict__ bv,
                                float* __restrict__ q, float* __restrict__ k,
                                float* __restrict__ v, int N) {
    int wave  = threadIdx.x >> 5;
    int lane  = threadIdx.x & 31;
    int khalf = lane >> 4;
    int l15   = lane & 15;
    int mBase = blockIdx.x * 16;
    int cBase = wave * 16;
    int which = blockIdx.z;

    const _Float16* W = w16qkv + (size_t)which * CCH * CCH;
    const float* bias = (which == 0) ? bq : ((which == 1) ? bk : bv);
    float* out        = (which == 0) ? q  : ((which == 1) ? k  : v);

    v8f acc = {};
#pragma unroll
    for (int kb = 0; kb < CCH; kb += 32) {
        v16h a = load_frag_f16(x16, mBase + l15, kb, khalf);
        v16h b = load_frag_f16(W,   cBase + l15, kb, khalf);
        acc = __builtin_amdgcn_wmma_f32_16x16x32_f16(false, a, false, b, (short)0, acc,
                                                     false, false);
    }
    int col = cBase + l15;
    float bsv = bias[col];
    float* dst = out + (size_t)(mBase + 8 * khalf) * CCH + col;
    if (mBase + 16 <= N) {       // full tile: branch-free stores
#pragma unroll
        for (int r = 0; r < 8; ++r) dst[(size_t)r * CCH] = acc[r] + bsv;
    } else {
#pragma unroll
        for (int r = 0; r < 8; ++r) {
            int m = mBase + r + 8 * khalf;
            if (m < N) out[(size_t)m * CCH + col] = acc[r] + bsv;
        }
    }
}

// ---------------- output projection GEMM (acc*S @ Wo.T)/S + bo via WMMA ----------------
__global__ void gemm_out_kernel(const float* __restrict__ accin,
                                const _Float16* __restrict__ wo16,
                                const float* __restrict__ bo, float* __restrict__ out, int N) {
    int wave  = threadIdx.x >> 5;
    int lane  = threadIdx.x & 31;
    int khalf = lane >> 4;
    int l15   = lane & 15;
    int mBase = blockIdx.x * 16;
    int cBase = wave * 16;

    v8f acc = {};
#pragma unroll
    for (int kb = 0; kb < CCH; kb += 32) {
        v16h a = load_frag_f32(accin, mBase + l15, kb, khalf, OUT_SCALE);
        v16h b = load_frag_f16(wo16,  cBase + l15, kb, khalf);
        acc = __builtin_amdgcn_wmma_f32_16x16x32_f16(false, a, false, b, (short)0, acc,
                                                     false, false);
    }
    int col = cBase + l15;
    float bsv = bo[col];
    float* dst = out + (size_t)(mBase + 8 * khalf) * CCH + col;
    if (mBase + 16 <= N) {
#pragma unroll
        for (int r = 0; r < 8; ++r) dst[(size_t)r * CCH] = acc[r] * OUT_INV_SCALE + bsv;
    } else {
#pragma unroll
        for (int r = 0; r < 8; ++r) {
            int m = mBase + r + 8 * khalf;
            if (m < N) out[(size_t)m * CCH + col] = acc[r] * OUT_INV_SCALE + bsv;
        }
    }
}

// ---------------- per-edge attention scores ----------------
// thread t -> edge e = t>>3, head h = t&7
__global__ void edge_scores_kernel(const int* __restrict__ ei, const float* __restrict__ ew,
                                   const float* __restrict__ q, const float* __restrict__ k,
                                   const float* __restrict__ xm, const float* __restrict__ Wd,
                                   const float* __restrict__ bd, float* __restrict__ scores,
                                   int E) {
    int t = blockIdx.x * blockDim.x + threadIdx.x;
    int e = t >> 3;
    int h = t & 7;
    if (e >= E) return;
    int r = ei[e];
    int c = ei[E + e];
    const float4* qp = (const float4*)(q + (size_t)r * CCH + h * HD);
    const float4* kp = (const float4*)(k + (size_t)c * CCH + h * HD);
    float s = 0.0f;
#pragma unroll
    for (int i = 0; i < 4; ++i) {
        float4 a = qp[i], b = kp[i];
        s += a.x * b.x + a.y * b.y + a.z * b.z + a.w * b.w;
    }
    s *= 0.25f;                       // 1/sqrt(HD)
    s += ew[e];
    s += (xm[c] - xm[r]) * Wd[h] + bd[h];
    scores[(size_t)e * HEADS + h] = s;
}

// ---------------- softmax over edge dim, per head ----------------
__global__ void init_stats_kernel(float* __restrict__ maxv, float* __restrict__ sums) {
    int t = threadIdx.x;
    if (t < HEADS) { maxv[t] = -INFINITY; sums[t] = 0.0f; }
}

__device__ __forceinline__ void atomicMaxF(float* addr, float val) {
    int* ia = (int*)addr;
    int old = *ia;
    while (__int_as_float(old) < val) {
        int assumed = old;
        old = atomicCAS(ia, assumed, __float_as_int(val));
        if (old == assumed) break;
    }
}

// grid*block is a multiple of 8 -> each thread owns a fixed head h = tid&7
__global__ void reduce_max_kernel(const float* __restrict__ scores, float* __restrict__ maxv,
                                  int total) {
    int stride = gridDim.x * blockDim.x;
    float m = -INFINITY;
    for (int i = blockIdx.x * blockDim.x + threadIdx.x; i < total; i += stride)
        m = fmaxf(m, scores[i]);
    m = fmaxf(m, __shfl_xor(m, 8));
    m = fmaxf(m, __shfl_xor(m, 16));
    int lane = threadIdx.x & 31;
    if (lane < 8) atomicMaxF(&maxv[lane], m);
}

__global__ void exp_sum_kernel(float* __restrict__ scores, const float* __restrict__ maxv,
                               float* __restrict__ sums, int total) {
    int stride = gridDim.x * blockDim.x;
    int tid = blockIdx.x * blockDim.x + threadIdx.x;
    float mx = maxv[tid & 7];
    float s = 0.0f;
    for (int i = tid; i < total; i += stride) {
        float ev = expf(scores[i] - mx);
        scores[i] = ev;
        s += ev;
    }
    s += __shfl_xor(s, 8);
    s += __shfl_xor(s, 16);
    int lane = threadIdx.x & 31;
    if (lane < 8) unsafeAtomicAdd(&sums[lane], s);
}

__global__ void inv_stats_kernel(float* __restrict__ sums) {
    int t = threadIdx.x;
    if (t < HEADS) sums[t] = 1.0f / sums[t];
}

// ---------------- zero accumulator ----------------
__global__ void zero_kernel(float* __restrict__ p, int n) {
    int t = blockIdx.x * blockDim.x + threadIdx.x;
    if (t < n) p[t] = 0.0f;
}

// ---------------- weighted-message scatter-add ----------------
// thread t -> edge e = t>>5, channels [4*(t&31), 4*(t&31)+3] (all within one head group)
__global__ void scatter_kernel(const int* __restrict__ ei, const float* __restrict__ vbuf,
                               const float* __restrict__ scores,
                               const float* __restrict__ inv_sums,
                               float* __restrict__ acc, int E) {
    unsigned int t = blockIdx.x * blockDim.x + threadIdx.x;
    unsigned int e = t >> 5;
    if (e >= (unsigned int)E) return;
    unsigned int cg = (t & 31u) * 4u;   // 0..124, 4 channels, same head (16 | cg group)
    int r  = ei[e];
    int cl = ei[E + e];
    int h  = (int)(cg >> 4);
    float attn = scores[(size_t)e * HEADS + h] * inv_sums[h];
    float4 vv = *(const float4*)(vbuf + (size_t)cl * CCH + cg);
    float* dst = acc + (size_t)r * CCH + cg;
    unsafeAtomicAdd(dst + 0, vv.x * attn);
    unsafeAtomicAdd(dst + 1, vv.y * attn);
    unsafeAtomicAdd(dst + 2, vv.z * attn);
    unsafeAtomicAdd(dst + 3, vv.w * attn);
}

// ---------------- host-side launch ----------------
extern "C" void kernel_launch(void* const* d_in, const int* in_sizes, int n_in,
                              void* d_out, int out_size, void* d_ws, size_t ws_size,
                              hipStream_t stream) {
    const float* x  = (const float*)d_in[0];
    const int*   ei = (const int*)d_in[1];
    const float* ew = (const float*)d_in[2];
    const float* Wq = (const float*)d_in[3];
    const float* bq = (const float*)d_in[4];
    const float* Wk = (const float*)d_in[5];
    const float* bk = (const float*)d_in[6];
    const float* Wv = (const float*)d_in[7];
    const float* bv = (const float*)d_in[8];
    const float* Wd = (const float*)d_in[9];
    const float* bd = (const float*)d_in[10];
    const float* Wo = (const float*)d_in[11];
    const float* bo = (const float*)d_in[12];
    float* out = (float*)d_out;

    int N = in_sizes[0] / CCH;
    int E = in_sizes[2];
    int Npad = (N + 15) & ~15;

    char* p = (char*)d_ws;
    auto walloc = [&](size_t bytes) -> char* {
        char* r = p;
        p += (bytes + 255) & ~(size_t)255;
        return r;
    };
    _Float16* x16    = (_Float16*)walloc((size_t)Npad * CCH * sizeof(_Float16));
    _Float16* w16qkv = (_Float16*)walloc((size_t)3 * CCH * CCH * sizeof(_Float16));
    _Float16* wo16   = (_Float16*)walloc((size_t)CCH * CCH * sizeof(_Float16));
    float* xm     = (float*)walloc((size_t)N * sizeof(float));
    float* qb     = (float*)walloc((size_t)Npad * CCH * sizeof(float)); // reused as acc
    float* kb     = (float*)walloc((size_t)Npad * CCH * sizeof(float));
    float* vb     = (float*)walloc((size_t)Npad * CCH * sizeof(float));
    float* scores = (float*)walloc((size_t)E * HEADS * sizeof(float));
    float* stats  = (float*)walloc(64);
    float* maxv = stats;
    float* sums = stats + 8;

    // 1) fp16 copy of x + per-node channel means
    prep_x_kernel<<<N, CCH, 0, stream>>>(x, x16, xm, N);
    // 2) fp16 weights
    conv_w_kernel<<<(4 * CCH * CCH + 255) / 256, 256, 0, stream>>>(Wq, Wk, Wv, Wo, w16qkv, wo16);
    // 3) Q/K/V projections (WMMA)
    gemm_qkv_kernel<<<dim3(Npad / 16, 1, 3), 256, 0, stream>>>(x16, w16qkv, bq, bk, bv,
                                                               qb, kb, vb, N);
    // 4) edge scores
    init_stats_kernel<<<1, 32, 0, stream>>>(maxv, sums);
    edge_scores_kernel<<<(E * HEADS + 255) / 256, 256, 0, stream>>>(ei, ew, qb, kb, xm,
                                                                    Wd, bd, scores, E);
    // 5) softmax over the edge dimension, per head
    reduce_max_kernel<<<1024, 256, 0, stream>>>(scores, maxv, E * HEADS);
    exp_sum_kernel<<<1024, 256, 0, stream>>>(scores, maxv, sums, E * HEADS);
    inv_stats_kernel<<<1, 32, 0, stream>>>(sums);
    // 6) scatter-add messages (acc aliases qb; q/k are dead now)
    float* acc = qb;
    zero_kernel<<<(Npad * CCH + 255) / 256, 256, 0, stream>>>(acc, Npad * CCH);
    scatter_kernel<<<((unsigned)E * 32u + 255) / 256, 256, 0, stream>>>(ei, vb, scores,
                                                                        sums, acc, E);
    // 7) output projection (WMMA, scaled through fp16)
    gemm_out_kernel<<<dim3(Npad / 16, 1, 1), 256, 0, stream>>>(acc, wo16, bo, out, N);
}
